// TSPModel_26757646254311
// MI455X (gfx1250) — compile-verified, matrix operand order checked by reading, not run on hardware
//
#include <hip/hip_runtime.h>
#include <hip/hip_bf16.h>

// ---------------------------------------------------------------------------
// TSP pointer-network forward for MI455X (gfx1250, wave32, WMMA bf16).
//   B=128, N=512, DIM=256, ENC_L=2, DEC_L=4, KNN=8
// All dense GEMMs use v_wmma_f32_16x16x32_bf16 with fp32 accumulation.
// Weights are pre-packed into per-lane fragment order (2 x global_load_b128
// per B fragment). Decoder: one persistent kernel, 8 WGs x 16 waves, one
// wave per batch row -> all per-row reductions are wave32 shuffles, only
// ~14 block barriers per step instead of ~100.
// ---------------------------------------------------------------------------

typedef __bf16 bf16_t;
typedef __bf16  v16bf __attribute__((ext_vector_type(16)));
typedef __bf16  v8bf  __attribute__((ext_vector_type(8)));
typedef float   v8f   __attribute__((ext_vector_type(8)));

#define KB_  128   // batch
#define KN_  512   // nodes
#define KD_  256   // dim

static __device__ __forceinline__ v8f wmma_bf16(v16bf a, v16bf b, v8f c) {
  // (neg_a, A, neg_b, B, c_mod, C, reuse_a, reuse_b)
  return __builtin_amdgcn_wmma_f32_16x16x32_bf16(false, a, false, b, (short)0, c, false, false);
}

// 16-bit A-matrix 16x32 per-lane K index for half h (0..15)  [ISA 7.12.2]
static __device__ __forceinline__ int kmap(int lane, int h) {
  return ((h & 8) << 1) + ((lane & 16) >> 1) + (h & 7);
}

static __device__ __forceinline__ float bf2f(unsigned u) {   // bf16 bits -> f32
  return __uint_as_float(u << 16);
}

static __device__ __forceinline__ float gumbel_noise(unsigned s) { // hash noise
  s ^= s >> 17; s *= 0xed5ad4bbu;
  s ^= s >> 11; s *= 0xac4c1b51u;
  s ^= s >> 15; s *= 0x31848babu;
  s ^= s >> 14;
  float u = (float)(s >> 8) * (1.0f / 16777216.0f) + 1e-7f;
  return -__logf(-__logf(u));
}

// A fragment (16x32 bf16 tile) from an fp32 row pointer: halves 0..7 are
// K = k0+kb8+0..7 (contiguous), halves 8..15 are K = k0+16+kb8+0..7.
static __device__ __forceinline__ v16bf load_afrag(const float* rowp, int k0, int lane) {
  int kb8 = (lane & 16) >> 1;                       // 0 or 8
  const float4* p0 = (const float4*)(rowp + k0 + kb8);
  const float4* p1 = (const float4*)(rowp + k0 + 16 + kb8);
  float4 a0 = p0[0], a1 = p0[1], a2 = p1[0], a3 = p1[1];
  v16bf a;
  a[0]  = (bf16_t)a0.x; a[1]  = (bf16_t)a0.y; a[2]  = (bf16_t)a0.z; a[3]  = (bf16_t)a0.w;
  a[4]  = (bf16_t)a1.x; a[5]  = (bf16_t)a1.y; a[6]  = (bf16_t)a1.z; a[7]  = (bf16_t)a1.w;
  a[8]  = (bf16_t)a2.x; a[9]  = (bf16_t)a2.y; a[10] = (bf16_t)a2.z; a[11] = (bf16_t)a2.w;
  a[12] = (bf16_t)a3.x; a[13] = (bf16_t)a3.y; a[14] = (bf16_t)a3.z; a[15] = (bf16_t)a3.w;
  return a;
}

// B fragment from pre-packed weights: 16 contiguous bf16 per lane (2 x b128).
static __device__ __forceinline__ v16bf load_wfrag(const bf16_t* __restrict__ P,
                                                   int nt, int kb, int K32, int lane) {
  const v8bf* p = (const v8bf*)(P + (((size_t)(nt * K32 + kb) * 32 + lane) << 4));
  union { v16bf v; v8bf h2[2]; } u;
  u.h2[0] = p[0];
  u.h2[1] = p[1];
  return u.v;
}

static __device__ __forceinline__ float blkSum256(float v, float* red) {
  int t = threadIdx.x;
  red[t] = v; __syncthreads();
  for (int s = 128; s > 0; s >>= 1) { if (t < s) red[t] += red[t + s]; __syncthreads(); }
  v = red[0]; __syncthreads();
  return v;
}

// ---------------------------------------------------------------------------
// fp32 -> bf16 conversion (plain layout, used for node_emb)
// ---------------------------------------------------------------------------
__global__ void f32_to_bf16_kernel(const float* __restrict__ in, bf16_t* __restrict__ out, int n) {
  int i = blockIdx.x * 256 + threadIdx.x;
  if (i < n) out[i] = (bf16_t)in[i];
}

// ---------------------------------------------------------------------------
// Pack W_f32[K,Nc] row-major into per-lane WMMA B-fragment order:
//   P[((nt*K32 + kb)*32 + lane)*16 + h] = (bf16) W[(kb*32+kmap(lane,h))*Nc + nt*16+(lane&15)]
// ---------------------------------------------------------------------------
__global__ void pack_w_kernel(const float* __restrict__ W, bf16_t* __restrict__ P,
                              int K, int Nc) {
  int gid = blockIdx.x * 256 + threadIdx.x;
  if (gid >= K * Nc) return;
  int K32 = K >> 5;
  int h    = gid & 15;
  int lane = (gid >> 4) & 31;
  int rest = gid >> 9;
  int kb   = rest % K32;
  int nt   = rest / K32;
  int k = kb * 32 + kmap(lane, h);
  int n = nt * 16 + (lane & 15);
  P[gid] = (bf16_t)W[k * Nc + n];
}

// ---------------------------------------------------------------------------
// h = coords @ enc_in_w + enc_in_b   (K=2, pure VALU)
// ---------------------------------------------------------------------------
__global__ void input_proj_kernel(const float* __restrict__ coords, const float* __restrict__ w,
                                  const float* __restrict__ bias, float* __restrict__ h) {
  int gid = blockIdx.x * 256 + threadIdx.x;   // over (B*N)*256
  int m = gid >> 8, d = gid & 255;
  h[gid] = coords[m * 2] * w[d] + coords[m * 2 + 1] * w[256 + d] + bias[d];
}

// ---------------------------------------------------------------------------
// GEMM: C[M,Nc] = A_f32[M,K] @ Wpacked_bf16[K,Nc] + bias (+res)
// one wave = one 16-row tile x TWO adjacent 16-col tiles (A fragment reuse).
// ---------------------------------------------------------------------------
__global__ void gemm_bf16_kernel(const float* __restrict__ A, const bf16_t* __restrict__ P,
                                 const float* __restrict__ bias, const float* __restrict__ res,
                                 float* __restrict__ C, int M, int Nc, int K, int tiles2) {
  int wave = threadIdx.x >> 5;
  int lane = threadIdx.x & 31;
  int t2 = blockIdx.x * 8 + wave;
  if (t2 >= tiles2) return;                   // wave-uniform
  int ntn2 = Nc >> 5;
  int K32 = K >> 5;
  int mt = t2 / ntn2, ntp = t2 % ntn2;
  int nt0 = 2 * ntp, nt1 = nt0 + 1;
  const float* arow = A + (size_t)(mt * 16 + (lane & 15)) * K;
  v8f c0 = {}, c1 = {};
#pragma unroll 4
  for (int kb = 0; kb < K32; ++kb) {
    v16bf a = load_afrag(arow, kb * 32, lane);
    c0 = wmma_bf16(a, load_wfrag(P, nt0, kb, K32, lane), c0);
    c1 = wmma_bf16(a, load_wfrag(P, nt1, kb, K32, lane), c1);
  }
  int rbase = (lane & 16) ? 8 : 0;
  int cn0 = nt0 * 16 + (lane & 15), cn1 = cn0 + 16;
  float bi0 = bias[cn0], bi1 = bias[cn1];
  if (res) {
#pragma unroll
    for (int i = 0; i < 8; ++i) {
      int idx0 = (mt * 16 + rbase + i) * Nc + cn0;
      int idx1 = idx0 + 16;
      C[idx0] = c0[i] + bi0 + res[idx0];
      C[idx1] = c1[i] + bi1 + res[idx1];
    }
  } else {
#pragma unroll
    for (int i = 0; i < 8; ++i) {
      int idx0 = (mt * 16 + rbase + i) * Nc + cn0;
      int idx1 = idx0 + 16;
      C[idx0] = c0[i] + bi0;
      C[idx1] = c1[i] + bi1;
    }
  }
}

// ---------------------------------------------------------------------------
// kn = k / max(||k||, 1e-12), k = qkv[:, 256:512]
// ---------------------------------------------------------------------------
__global__ void knorm_kernel(const float* __restrict__ qkv, float* __restrict__ kn) {
  __shared__ float red[256];
  int row = blockIdx.x;                       // b*512+n
  int t = threadIdx.x;
  float v = qkv[(size_t)row * 768 + 256 + t];
  float ss = blkSum256(v * v, red);
  float s = 1.f / fmaxf(sqrtf(ss), 1e-12f);
  kn[(size_t)row * 256 + t] = v * s;
}

// ---------------------------------------------------------------------------
// fused sim row + top-8 selection (diagonal masked)
// ---------------------------------------------------------------------------
__global__ void knn_kernel(const float* __restrict__ kn, int* __restrict__ idxout) {
  __shared__ __align__(16) float knrow[256];
  __shared__ float sims[512];
  __shared__ float red[256];
  __shared__ int redi[256];
  int row = blockIdx.x;
  int b = row >> 9, n = row & 511;
  int t = threadIdx.x;
  knrow[t] = kn[(size_t)row * 256 + t];
  __syncthreads();
  const float4* kr4 = (const float4*)knrow;
  for (int mm = t; mm < 512; mm += 256) {
    float acc;
    if (mm == n) acc = -1e9f;
    else {
      acc = 0.f;
      const float4* kp4 = (const float4*)(kn + ((size_t)b * 512 + mm) * 256);
#pragma unroll 8
      for (int d = 0; d < 64; ++d) {
        float4 x = kr4[d], y = kp4[d];
        acc += x.x * y.x + x.y * y.y + x.z * y.z + x.w * y.w;
      }
    }
    sims[mm] = acc;
  }
  __syncthreads();
  for (int it = 0; it < 8; ++it) {
    float bv = sims[t]; int bi = t;
    float ov = sims[t + 256];
    if (ov > bv) { bv = ov; bi = t + 256; }
    red[t] = bv; redi[t] = bi; __syncthreads();
    for (int s = 128; s > 0; s >>= 1) {
      if (t < s) {
        float o = red[t + s]; int oi = redi[t + s];
        if (o > red[t] || (o == red[t] && oi < redi[t])) { red[t] = o; redi[t] = oi; }
      }
      __syncthreads();
    }
    if (t == 0) { idxout[row * 8 + it] = redi[0]; sims[redi[0]] = -3e9f; }
    __syncthreads();
  }
}

// ---------------------------------------------------------------------------
// gather-attend over 8 neighbors: out = softmax(q . k_sel / 16) @ v_sel
// ---------------------------------------------------------------------------
__global__ void attend_kernel(const float* __restrict__ qkv, const int* __restrict__ idxin,
                              float* __restrict__ outp) {
  __shared__ float qrow[256];
  __shared__ float ks[8][256];
  __shared__ float vs[8][256];
  __shared__ float red[256];
  __shared__ float sc[8];
  __shared__ float w8[8];
  int row = blockIdx.x;
  int b = row >> 9;
  int t = threadIdx.x;
  qrow[t] = qkv[(size_t)row * 768 + t];
  for (int i = 0; i < 8; ++i) {
    int id = idxin[row * 8 + i];
    size_t base = ((size_t)b * 512 + id) * 768;
    ks[i][t] = qkv[base + 256 + t];
    vs[i][t] = qkv[base + 512 + t];
  }
  __syncthreads();
  for (int i = 0; i < 8; ++i) {
    float s = blkSum256(qrow[t] * ks[i][t], red);
    if (t == 0) sc[i] = s * (1.f / 16.f);     // 1/sqrt(256)
    __syncthreads();
  }
  if (t == 0) {
    float m = sc[0];
    for (int i = 1; i < 8; ++i) m = fmaxf(m, sc[i]);
    float sum = 0.f;
    for (int i = 0; i < 8; ++i) { w8[i] = __expf(sc[i] - m); sum += w8[i]; }
    float inv = 1.f / sum;
    for (int i = 0; i < 8; ++i) w8[i] *= inv;
  }
  __syncthreads();
  float acc = 0.f;
  for (int i = 0; i < 8; ++i) acc += w8[i] * vs[i][t];
  outp[(size_t)row * 256 + t] = acc;
}

// ---------------------------------------------------------------------------
// Persistent decoder: 8 blocks x 512 threads (16 waves), one wave per batch
// row, 511 sequential steps. Per-row math is wave32-shuffle based; block
// barriers only at the cross-wave WMMA GEMM phase boundaries.
// ---------------------------------------------------------------------------
__global__ __launch_bounds__(512) void tsp_decoder_kernel(
    const float* __restrict__ ne_f32, const bf16_t* __restrict__ ne_bf,
    const float* __restrict__ coords,
    const float* __restrict__ ln_g, const float* __restrict__ ln_b,
    const bf16_t* __restrict__ w1p, const float* __restrict__ b1p,
    const bf16_t* __restrict__ w2p, const float* __restrict__ b2p,
    const float* __restrict__ Avec, const float* __restrict__ Bvec,
    const bf16_t* __restrict__ qwp, const float* __restrict__ qbp,
    float* __restrict__ outp) {
  __shared__ __align__(16) float sX[16 * 256];   // running token x
  __shared__ __align__(16) float sB[16 * 256];   // hn (layers) / q (logits)
  __shared__ __align__(16) float sT[16 * 512];   // silu(hn@w1+b1) / masked logits
  __shared__ __align__(16) float sY[16 * 256];   // layer output y

  const int tid = threadIdx.x;
  const int wv = tid >> 5;                   // wave index == row r
  const int lane = tid & 31;
  const int brow = blockIdx.x * 16 + wv;     // batch index

  float st[32];                              // SSM states: 4 layers x 8 dims
#pragma unroll
  for (int i = 0; i < 32; ++i) st[i] = 0.f;

  unsigned vismask = (lane == 0) ? 1u : 0u;  // per-lane visited bits (node n: lane n&31, bit n>>5)
  int cur = 0;
  float lp = 0.f, ent = 0.f, tl = 0.f;       // lane 0 authoritative

  // init token = node_emb[b, 0]
#pragma unroll
  for (int kk = 0; kk < 8; ++kk) {
    int d = lane + 32 * kk;
    sX[wv * 256 + d] = ne_f32[((size_t)brow * 512 + 0) * 256 + d];
  }
  __syncthreads();

  for (int step = 0; step < KN_ - 1; ++step) {
    // ---------------- SSM stack (4 layers) ----------------
    for (int l = 0; l < 4; ++l) {
      __builtin_prefetch((const void*)(w1p + (size_t)l * 131072 + (size_t)wv * 8192), 0, 2);
      __builtin_prefetch((const void*)(w2p + (size_t)l * 131072 + (size_t)wv * 8192), 0, 2);

      // LayerNorm row wv: wave-local shuffle reductions
      float xv[8];
      float s = 0.f;
#pragma unroll
      for (int kk = 0; kk < 8; ++kk) { xv[kk] = sX[wv * 256 + lane + 32 * kk]; s += xv[kk]; }
#pragma unroll
      for (int m = 16; m > 0; m >>= 1) s += __shfl_xor(s, m, 32);
      float mean = s * (1.f / 256.f);
      float vv = 0.f;
#pragma unroll
      for (int kk = 0; kk < 8; ++kk) { float d0 = xv[kk] - mean; vv += d0 * d0; }
#pragma unroll
      for (int m = 16; m > 0; m >>= 1) vv += __shfl_xor(vv, m, 32);
      float rstd = rsqrtf(vv * (1.f / 256.f) + 1e-5f);
#pragma unroll
      for (int kk = 0; kk < 8; ++kk) {
        int d = lane + 32 * kk;
        sB[wv * 256 + d] = (xv[kk] - mean) * rstd * ln_g[l * 256 + d] + ln_b[l * 256 + d];
      }
      __syncthreads();

      // GEMM1: [16x256] @ w1[256x512] -> silu -> sT  (wave wv -> n-tiles 2wv, 2wv+1)
      {
        const bf16_t* P = w1p + (size_t)l * 131072;
        const float* arow = &sB[(lane & 15) * 256];
        v8f c0 = {}, c1 = {};
#pragma unroll
        for (int kb = 0; kb < 8; ++kb) {
          v16bf a = load_afrag(arow, kb * 32, lane);
          c0 = wmma_bf16(a, load_wfrag(P, 2 * wv, kb, 8, lane), c0);
          c1 = wmma_bf16(a, load_wfrag(P, 2 * wv + 1, kb, 8, lane), c1);
        }
        int rbase = (lane & 16) ? 8 : 0;
        int cn0 = (2 * wv) * 16 + (lane & 15), cn1 = cn0 + 16;
        float bi0 = b1p[l * 512 + cn0], bi1 = b1p[l * 512 + cn1];
#pragma unroll
        for (int i = 0; i < 8; ++i) {
          float v0 = c0[i] + bi0; v0 = v0 / (1.f + __expf(-v0));
          float v1 = c1[i] + bi1; v1 = v1 / (1.f + __expf(-v1));
          sT[(rbase + i) * 512 + cn0] = v0;
          sT[(rbase + i) * 512 + cn1] = v1;
        }
      }
      __syncthreads();

      // GEMM2: [16x512] @ w2[512x256] -> sY  (wave wv -> n-tile wv)
      {
        const bf16_t* P = w2p + (size_t)l * 131072;
        const float* arow = &sT[(lane & 15) * 512];
        v8f c = {};
#pragma unroll 4
        for (int kb = 0; kb < 16; ++kb) {
          v16bf a = load_afrag(arow, kb * 32, lane);
          c = wmma_bf16(a, load_wfrag(P, wv, kb, 16, lane), c);
        }
        int rbase = (lane & 16) ? 8 : 0;
        int cn = wv * 16 + (lane & 15);
        float bi = b2p[l * 256 + cn];
#pragma unroll
        for (int i = 0; i < 8; ++i)
          sY[(rbase + i) * 256 + cn] = c[i] + bi;
      }
      __syncthreads();

      // state update + residual (wave-local: row wv)
#pragma unroll
      for (int kk = 0; kk < 8; ++kk) {
        int d = lane + 32 * kk;
        float y = sY[wv * 256 + d];
        st[l * 8 + kk] = tanhf(Avec[l * 256 + d] * st[l * 8 + kk] + Bvec[l * 256 + d] * y);
        sX[wv * 256 + d] += y;
      }
      // no barrier: next LN reads only this wave's row of sX
    }
    __syncthreads();                         // sX complete for cross-wave q-proj

    // ---------------- q projection: sX @ qw + qb -> sB (wave wv -> tile wv) --
    {
      const float* arow = &sX[(lane & 15) * 256];
      v8f c = {};
#pragma unroll
      for (int kb = 0; kb < 8; ++kb) {
        v16bf a = load_afrag(arow, kb * 32, lane);
        c = wmma_bf16(a, load_wfrag(qwp, wv, kb, 8, lane), c);
      }
      int rbase = (lane & 16) ? 8 : 0;
      int cn = wv * 16 + (lane & 15);
      float bi = qbp[cn];
#pragma unroll
      for (int i = 0; i < 8; ++i)
        sB[(rbase + i) * 256 + cn] = c[i] + bi;
    }
    __syncthreads();

    // ---------------- logits + mask + gumbel argmax (all wave-local) --------
    float amv = -3.4e38f; int ami = 0;
    float pmax = -3.4e38f;
    const float4* qf = (const float4*)&sB[wv * 256];
#pragma unroll 2
    for (int kk = 0; kk < 16; ++kk) {
      int n = lane + 32 * kk;
      const uint4* pk = (const uint4*)(ne_bf + ((size_t)brow * 512 + n) * 256);
      float acc = 0.f;
#pragma unroll 8
      for (int i = 0; i < 32; ++i) {
        uint4 u = pk[i];
        float4 q0 = qf[2 * i], q1 = qf[2 * i + 1];
        acc += q0.x * bf2f(u.x & 0xffffu) + q0.y * bf2f(u.x >> 16);
        acc += q0.z * bf2f(u.y & 0xffffu) + q0.w * bf2f(u.y >> 16);
        acc += q1.x * bf2f(u.z & 0xffffu) + q1.y * bf2f(u.z >> 16);
        acc += q1.z * bf2f(u.w & 0xffffu) + q1.w * bf2f(u.w >> 16);
      }
      float lm = ((vismask >> kk) & 1u) ? -1e9f : acc;
      sT[wv * 512 + n] = lm;
      pmax = fmaxf(pmax, lm);
      float gl = lm + gumbel_noise(((unsigned)step * 128u + (unsigned)brow) * 512u + (unsigned)n);
      if (gl > amv || (gl == amv && n < ami)) { amv = gl; ami = n; }
    }
#pragma unroll
    for (int m = 16; m > 0; m >>= 1) {
      float ov = __shfl_xor(amv, m, 32);
      int oi = __shfl_xor(ami, m, 32);
      if (ov > amv || (ov == amv && oi < ami)) { amv = ov; ami = oi; }
      pmax = fmaxf(pmax, __shfl_xor(pmax, m, 32));
    }
    int nxt = ami;

    float s1 = 0.f, s2 = 0.f;
#pragma unroll
    for (int kk = 0; kk < 16; ++kk) {
      float lm = sT[wv * 512 + lane + 32 * kk];
      float ex = __expf(lm - pmax);
      s1 += ex; s2 += ex * lm;
    }
#pragma unroll
    for (int m = 16; m > 0; m >>= 1) { s1 += __shfl_xor(s1, m, 32); s2 += __shfl_xor(s2, m, 32); }

    if (lane == (nxt & 31)) vismask |= 1u << (nxt >> 5);
    if (lane == 0) {
      float lse = pmax + __logf(s1);
      lp += sT[wv * 512 + nxt] - lse;                  // log-prob of chosen
      ent += lse - s2 / s1;                            // -sum p*logp
      float dx = coords[((size_t)brow * 512 + cur) * 2 + 0] - coords[((size_t)brow * 512 + nxt) * 2 + 0];
      float dy = coords[((size_t)brow * 512 + cur) * 2 + 1] - coords[((size_t)brow * 512 + nxt) * 2 + 1];
      tl += sqrtf(dx * dx + dy * dy);
    }
    cur = nxt;

    // token = node_emb[b, nxt] (wave-local)
#pragma unroll
    for (int kk = 0; kk < 8; ++kk) {
      int d = lane + 32 * kk;
      sX[wv * 256 + d] = ne_f32[((size_t)brow * 512 + nxt) * 256 + d];
    }
    __syncthreads();   // protect sB/sT reuse by next step's cross-wave GEMMs
  }

  if (lane == 0) {
    float dx = coords[((size_t)brow * 512 + cur) * 2 + 0] - coords[((size_t)brow * 512 + 0) * 2 + 0];
    float dy = coords[((size_t)brow * 512 + cur) * 2 + 1] - coords[((size_t)brow * 512 + 0) * 2 + 1];
    outp[brow] = lp;
    outp[KB_ + brow] = ent;
    outp[2 * KB_ + brow] = tl + sqrtf(dx * dx + dy * dy);
  }
}

// ---------------------------------------------------------------------------
// host launcher
// ---------------------------------------------------------------------------
extern "C" void kernel_launch(void* const* d_in, const int* in_sizes, int n_in,
                              void* d_out, int out_size, void* d_ws, size_t ws_size,
                              hipStream_t stream) {
  (void)in_sizes; (void)n_in; (void)out_size; (void)ws_size;
  const float* coords    = (const float*)d_in[0];
  const float* enc_in_w  = (const float*)d_in[1];
  const float* enc_in_b  = (const float*)d_in[2];
  const float* enc_qkv_w = (const float*)d_in[3];
  const float* enc_qkv_b = (const float*)d_in[4];
  const float* enc_out_w = (const float*)d_in[5];
  const float* enc_out_b = (const float*)d_in[6];
  const float* dec_ln_g  = (const float*)d_in[7];
  const float* dec_ln_b  = (const float*)d_in[8];
  const float* dec_w1    = (const float*)d_in[9];
  const float* dec_b1    = (const float*)d_in[10];
  const float* dec_w2    = (const float*)d_in[11];
  const float* dec_b2    = (const float*)d_in[12];
  const float* dec_A     = (const float*)d_in[13];
  const float* dec_B     = (const float*)d_in[14];
  const float* q_w       = (const float*)d_in[15];
  const float* q_b       = (const float*)d_in[16];
  float* out = (float*)d_out;

  // workspace layout (~421 MiB assumed available)
  char* ws = (char*)d_ws;
  size_t off = 0;
  auto alloc = [&](size_t bytes) -> void* {
    void* p = ws + off;
    off = (off + bytes + 255) & ~(size_t)255;
    return p;
  };
  const size_t MN = (size_t)KB_ * KN_;          // 65536 rows
  float* h     = (float*)alloc(MN * KD_ * 4);          // node features
  float* qkv   = (float*)alloc(MN * 3 * KD_ * 4);      // q|k|v
  float* kn    = (float*)alloc(MN * KD_ * 4);          // normalized k
  float* aout  = (float*)alloc(MN * KD_ * 4);          // attention out
  int*   idx   = (int*)  alloc(MN * 8 * 4);            // top-8 indices
  bf16_t* nebf = (bf16_t*)alloc(MN * KD_ * 2);         // node_emb bf16
  bf16_t* wqkv = (bf16_t*)alloc((size_t)2 * KD_ * 3 * KD_ * 2);  // packed
  bf16_t* wout = (bf16_t*)alloc((size_t)2 * KD_ * KD_ * 2);      // packed
  bf16_t* w1b  = (bf16_t*)alloc((size_t)4 * KD_ * 2 * KD_ * 2);  // packed
  bf16_t* w2b  = (bf16_t*)alloc((size_t)4 * 2 * KD_ * KD_ * 2);  // packed
  bf16_t* qwb  = (bf16_t*)alloc((size_t)KD_ * KD_ * 2);          // packed

  auto pack = [&](const float* src, bf16_t* dst, int K, int Nc) {
    int n = K * Nc;
    pack_w_kernel<<<(n + 255) / 256, 256, 0, stream>>>(src, dst, K, Nc);
  };
  for (int l = 0; l < 2; ++l) {
    pack(enc_qkv_w + (size_t)l * KD_ * 3 * KD_, wqkv + (size_t)l * KD_ * 3 * KD_, KD_, 3 * KD_);
    pack(enc_out_w + (size_t)l * KD_ * KD_,     wout + (size_t)l * KD_ * KD_,     KD_, KD_);
  }
  for (int l = 0; l < 4; ++l) {
    pack(dec_w1 + (size_t)l * KD_ * 2 * KD_, w1b + (size_t)l * KD_ * 2 * KD_, KD_, 2 * KD_);
    pack(dec_w2 + (size_t)l * 2 * KD_ * KD_, w2b + (size_t)l * 2 * KD_ * KD_, 2 * KD_, KD_);
  }
  pack(q_w, qwb, KD_, KD_);

  // encoder input projection
  input_proj_kernel<<<(unsigned)(MN * KD_ / 256), 256, 0, stream>>>(coords, enc_in_w, enc_in_b, h);

  // encoder attention layers
  for (int l = 0; l < 2; ++l) {
    {
      int M = (int)MN, Nc = 3 * KD_, K = KD_;
      int tiles2 = (M / 16) * (Nc / 32);
      gemm_bf16_kernel<<<tiles2 / 8, 256, 0, stream>>>(
          h, wqkv + (size_t)l * KD_ * 3 * KD_, enc_qkv_b + l * 3 * KD_, nullptr,
          qkv, M, Nc, K, tiles2);
    }
    knorm_kernel<<<(unsigned)MN, 256, 0, stream>>>(qkv, kn);
    knn_kernel<<<(unsigned)MN, 256, 0, stream>>>(kn, idx);
    attend_kernel<<<(unsigned)MN, 256, 0, stream>>>(qkv, idx, aout);
    {
      int M = (int)MN, Nc = KD_, K = KD_;
      int tiles2 = (M / 16) * (Nc / 32);
      gemm_bf16_kernel<<<tiles2 / 8, 256, 0, stream>>>(
          aout, wout + (size_t)l * KD_ * KD_, enc_out_b + l * KD_, h,
          h, M, Nc, K, tiles2);
    }
  }

  // node_emb bf16 copy (keeps decoder logits traffic L2-resident)
  f32_to_bf16_kernel<<<(unsigned)((MN * KD_ + 255) / 256), 256, 0, stream>>>(h, nebf, (int)(MN * KD_));

  // persistent decoder: 8 WGs x 16 waves (one wave per batch row)
  tsp_decoder_kernel<<<8, 512, 0, stream>>>(
      h, nebf, coords, dec_ln_g, dec_ln_b, w1b, dec_b1, w2b, dec_b2,
      dec_A, dec_B, qwb, q_b, out);
}
